// EMAQuantizer_91130616086753
// MI455X (gfx1250) — compile-verified, matrix operand order checked by reading, not run on hardware
//
#include <hip/hip_runtime.h>
#include <hip/hip_bf16.h>

typedef __attribute__((ext_vector_type(16))) __bf16 v16bf;
typedef __attribute__((ext_vector_type(8)))  float  v8f;

#define K_CODES 1024
#define DIM     128
#define N_ROWS  65536
#define DECAYF  0.99f
#define EPSF    1e-5f
#define CCOST   0.25f

// ---------------------------------------------------------------------------
// Zero atomic-accumulation scratch (counts | embed_sum | sse are contiguous)
// ---------------------------------------------------------------------------
__global__ void vq_zero(float* __restrict__ p, int n) {
    int i = blockIdx.x * blockDim.x + threadIdx.x;
    if (i < n) p[i] = 0.0f;
}

// ---------------------------------------------------------------------------
// Split codebook into bf16 hi/lo (code-major [K][D]), f32 transpose, ||e||^2
// ---------------------------------------------------------------------------
__global__ void vq_precompute(const float* __restrict__ embed,   // [D][K]
                              float* __restrict__ eT,            // [K][D]
                              __bf16* __restrict__ ehi,          // [K][D]
                              __bf16* __restrict__ elo,          // [K][D]
                              float* __restrict__ enorm) {       // [K]
    const int k = blockIdx.x;      // 1024
    const int d = threadIdx.x;     // 128
    float v = embed[d * K_CODES + k];
    eT[k * DIM + d] = v;
    __bf16 h = (__bf16)v;
    float  r = v - (float)h;
    ehi[k * DIM + d] = h;
    elo[k * DIM + d] = (__bf16)r;

    __shared__ float red[DIM];
    red[d] = v * v;
    __syncthreads();
    for (int s = 64; s > 0; s >>= 1) {
        if (d < s) red[d] += red[d + s];
        __syncthreads();
    }
    if (d == 0) enorm[k] = red[0];
}

// ---------------------------------------------------------------------------
// Main kernel: split-bf16 WMMA distance GEMM + argmin + gather + EMA stats
// 8 waves / block, each wave owns one 16-row tile of x.
// Codebook streamed through LDS in 32-code super-tiles, double-buffered via
// GLOBAL_LOAD_ASYNC_TO_LDS (ASYNCcnt DMA) so fill latency hides under WMMA.
// ---------------------------------------------------------------------------
__global__ __launch_bounds__(256) void vq_main(
    const float*  __restrict__ x,        // [N][128]
    const __bf16* __restrict__ ehi,      // [K][128]
    const __bf16* __restrict__ elo,      // [K][128]
    const float*  __restrict__ eT,       // [K][128]
    const float*  __restrict__ enorm,    // [K]
    float* __restrict__ out_q,           // [N*128]
    float* __restrict__ out_idx,         // [N] (as float)
    float* __restrict__ counts,          // [K]
    float* __restrict__ embed_sum,       // [D][K]
    float* __restrict__ sse_acc)         // [1]
{
    // two buffers x (32 codes * 128 dims) x (hi,lo) = 2 * 16KB
    __shared__ __align__(16) __bf16 sh_hi[2][32 * DIM];
    __shared__ __align__(16) __bf16 sh_lo[2][32 * DIM];
    __shared__ int sh_idx[8][16];

    const int lane = threadIdx.x & 31;
    const int wave = threadIdx.x >> 5;
    const int tile = blockIdx.x * 8 + wave;   // 16-row tile id
    const int m    = lane & 15;
    const int half = lane >> 4;
    const int row  = tile * 16 + m;

    // ---- Build A fragments (hi + residual lo) for the 4 K-chunks of D=128.
    // 16-bit A layout: lane(0-15)=rows, VGPR0..3 = K(base0..+7) pairs,
    // VGPR4..7 = K(base1..+7); lanes 16-31 shift K by 8.
    v16bf ahi[4], alo[4];
#pragma unroll
    for (int c = 0; c < 4; ++c) {
        const float* src0 = x + (size_t)row * DIM + c * 32 + half * 8;
        const float* src1 = src0 + 16;
#pragma unroll
        for (int i = 0; i < 8; ++i) {
            float v0 = src0[i];
            float v1 = src1[i];
            __bf16 h0 = (__bf16)v0, h1 = (__bf16)v1;
            ahi[c][i]     = h0;
            ahi[c][8 + i] = h1;
            alo[c][i]     = (__bf16)(v0 - (float)h0);
            alo[c][8 + i] = (__bf16)(v1 - (float)h1);
        }
    }

    float minv[8];
    int   mini[8];
#pragma unroll
    for (int j = 0; j < 8; ++j) { minv[j] = 3.4e38f; mini[j] = 0; }

    // Async DMA fill of one super-tile buffer: 8KB hi + 8KB lo.
    // 256 threads x 16B x (2 hi + 2 lo) issues; each lane supplies its own
    // LDS byte address (low 32 bits of the flat shared pointer) and the
    // 64-bit global address.  Tracked with ASYNCcnt.
    const int tid16 = threadIdx.x * 16;
    auto async_fill = [&](int b, int kst) {
        const char* gh = (const char*)(ehi + kst * 32 * DIM) + tid16;
        const char* gl = (const char*)(elo + kst * 32 * DIM) + tid16;
        unsigned lh = (unsigned)(uintptr_t)(&sh_hi[b][0]) + tid16;
        unsigned ll = (unsigned)(uintptr_t)(&sh_lo[b][0]) + tid16;
#pragma unroll
        for (int h = 0; h < 2; ++h) {
            asm volatile("global_load_async_to_lds_b128 %0, %1, off"
                         :: "v"(lh + h * 4096), "v"(gh + h * 4096) : "memory");
            asm volatile("global_load_async_to_lds_b128 %0, %1, off"
                         :: "v"(ll + h * 4096), "v"(gl + h * 4096) : "memory");
        }
    };

    // prologue: fill buffer 0 with super-tile 0
    async_fill(0, 0);
    asm volatile("s_wait_asynccnt 0x0" ::: "memory");
    __syncthreads();

    // ---- Sweep the codebook: 32 super-tiles of 32 codes.
    for (int kst = 0; kst < K_CODES / 32; ++kst) {
        const int cur = kst & 1;
        if (kst + 1 < K_CODES / 32) async_fill(cur ^ 1, kst + 1);

#pragma unroll
        for (int t = 0; t < 2; ++t) {         // two 16-code column tiles
            v8f acc = {};
#pragma unroll
            for (int c = 0; c < 4; ++c) {
                union { v16bf v; uint4 u[2]; } bh, bl;
                const __bf16* ph = &sh_hi[cur][0] + (t * 16 + m) * DIM + c * 32 + half * 8;
                const __bf16* pl = &sh_lo[cur][0] + (t * 16 + m) * DIM + c * 32 + half * 8;
                bh.u[0] = *(const uint4*)ph;
                bh.u[1] = *(const uint4*)(ph + 16);
                bl.u[0] = *(const uint4*)pl;
                bl.u[1] = *(const uint4*)(pl + 16);
                // split product: x.e = xhi.ehi + xhi.elo + xlo.ehi
                acc = __builtin_amdgcn_wmma_f32_16x16x32_bf16(
                          false, ahi[c], false, bh.v, (short)0, acc, false, false);
                acc = __builtin_amdgcn_wmma_f32_16x16x32_bf16(
                          false, ahi[c], false, bl.v, (short)0, acc, false, false);
                acc = __builtin_amdgcn_wmma_f32_16x16x32_bf16(
                          false, alo[c], false, bh.v, (short)0, acc, false, false);
            }
            // C/D layout: this lane holds column (kst*32 + t*16 + m),
            // rows 8*half + j.  ||x||^2 is row-constant -> argmin over
            // (||e||^2 - 2 x.e).
            const int   col = kst * 32 + t * 16 + m;
            const float en  = enorm[col];
#pragma unroll
            for (int j = 0; j < 8; ++j) {
                float val = en - 2.0f * acc[j];
                if (val < minv[j]) { minv[j] = val; mini[j] = col; }
            }
        }

        // next buffer landed + everyone done with current before overwrite
        asm volatile("s_wait_asynccnt 0x0" ::: "memory");
        __syncthreads();
    }

    // ---- min/argmin across the 16 lanes of each half-group
#pragma unroll
    for (int j = 0; j < 8; ++j) {
#pragma unroll
        for (int mask = 1; mask <= 8; mask <<= 1) {
            float ov = __shfl_xor(minv[j], mask, 32);
            int   oi = __shfl_xor(mini[j], mask, 32);
            if (ov < minv[j] || (ov == minv[j] && oi < mini[j])) {
                minv[j] = ov; mini[j] = oi;
            }
        }
    }
    if (m == 0) {
#pragma unroll
        for (int j = 0; j < 8; ++j) {
            int r = half * 8 + j;
            sh_idx[wave][r] = mini[j];
            out_idx[tile * 16 + r] = (float)mini[j];
            atomicAdd(&counts[mini[j]], 1.0f);
        }
    }

    // ---- gather quantized rows, loss partial, embed_sum segment-sum
    // lane -> row (lane&15), dims [half*64, half*64+64)
    const int    myidx = sh_idx[wave][m];
    const float* qsrc  = eT   + (size_t)myidx * DIM + half * 64;
    const float* xsrc  = x    + (size_t)row   * DIM + half * 64;
    float*       qdst  = out_q + (size_t)row  * DIM + half * 64;
    float s = 0.0f;
#pragma unroll 4
    for (int i = 0; i < 64; ++i) {
        float qv = qsrc[i];
        float xv = xsrc[i];
        qdst[i] = qv;
        float d = qv - xv;
        s += d * d;
        atomicAdd(&embed_sum[(half * 64 + i) * K_CODES + myidx], xv);
    }
#pragma unroll
    for (int mask = 1; mask < 32; mask <<= 1) s += __shfl_xor(s, mask, 32);
    if (lane == 0) atomicAdd(sse_acc, s);
}

// ---------------------------------------------------------------------------
// Finalize 1: normalized new_cluster_size (needs sum over K) + scalar loss
// ---------------------------------------------------------------------------
__global__ void vq_fin1(const float* __restrict__ cluster_size,
                        const float* __restrict__ counts,
                        const float* __restrict__ sse,
                        float* __restrict__ out_cs,
                        float* __restrict__ out_loss) {
    __shared__ float red[K_CODES];
    const int k = threadIdx.x;
    float raw = cluster_size[k] * DECAYF + (1.0f - DECAYF) * counts[k];
    red[k] = raw;
    __syncthreads();
    for (int s = 512; s > 0; s >>= 1) {
        if (k < s) red[k] += red[k + s];
        __syncthreads();
    }
    float n = red[0];
    out_cs[k] = (raw + EPSF) / (n + (float)K_CODES * EPSF) * n;
    if (k == 0) out_loss[0] = CCOST * sse[0] / (float)((size_t)N_ROWS * DIM);
}

// ---------------------------------------------------------------------------
// Finalize 2: EMA embed_avg update + new_embed = avg / cluster_size
// ---------------------------------------------------------------------------
__global__ void vq_fin2(const float* __restrict__ embed_avg,
                        const float* __restrict__ embed_sum,
                        const float* __restrict__ new_cs,
                        float* __restrict__ out_embed,
                        float* __restrict__ out_avg) {
    int i = blockIdx.x * blockDim.x + threadIdx.x;   // 131072
    float avg = embed_avg[i] * DECAYF + (1.0f - DECAYF) * embed_sum[i];
    out_avg[i]   = avg;
    out_embed[i] = avg / new_cs[i & (K_CODES - 1)];
}

// ---------------------------------------------------------------------------
extern "C" void kernel_launch(void* const* d_in, const int* in_sizes, int n_in,
                              void* d_out, int out_size, void* d_ws, size_t ws_size,
                              hipStream_t stream) {
    (void)in_sizes; (void)n_in; (void)out_size; (void)ws_size;

    const float* inputs       = (const float*)d_in[0];   // [64,1024,128]
    const float* embed        = (const float*)d_in[1];   // [128,1024]
    const float* cluster_size = (const float*)d_in[2];   // [1024]
    const float* embed_avg    = (const float*)d_in[3];   // [128,1024]

    char*   ws        = (char*)d_ws;
    float*  eT        = (float*) ws;                 // 524288 B
    __bf16* ehi       = (__bf16*)(ws + 524288);      // 262144 B
    __bf16* elo       = (__bf16*)(ws + 786432);      // 262144 B
    float*  enorm     = (float*)(ws + 1048576);      //   4096 B
    float*  counts    = (float*)(ws + 1052672);      //   4096 B
    float*  embed_sum = (float*)(ws + 1056768);      // 524288 B
    float*  sse       = (float*)(ws + 1581056);      //      4 B

    float* out       = (float*)d_out;
    float* out_q     = out;                 // 8388608
    float* out_loss  = out + 8388608;       // 1
    float* out_idx   = out + 8388609;       // 65536
    float* out_embed = out + 8454145;       // 131072
    float* out_cs    = out + 8585217;       // 1024
    float* out_avg   = out + 8586241;       // 131072

    // counts | embed_sum | sse are laid out contiguously: zero in one pass
    const int zn = K_CODES + DIM * K_CODES + 1;
    vq_zero<<<(zn + 255) / 256, 256, 0, stream>>>(counts, zn);

    vq_precompute<<<K_CODES, DIM, 0, stream>>>(embed, eT, ehi, elo, enorm);

    // 65536 rows / 16 per wave / 8 waves per block = 512 blocks
    vq_main<<<512, 256, 0, stream>>>(inputs, ehi, elo, eT, enorm,
                                     out_q, out_idx, counts, embed_sum, sse);

    vq_fin1<<<1, 1024, 0, stream>>>(cluster_size, counts, sse, out_cs, out_loss);
    vq_fin2<<<512, 256, 0, stream>>>(embed_avg, embed_sum, out_cs, out_embed, out_avg);
}
